// dendriticNet_76484777607574
// MI455X (gfx1250) — compile-verified, matrix operand order checked by reading, not run on hardware
//
#include <hip/hip_runtime.h>

// ---- CDNA5 (gfx1250) types for WMMA ----
typedef __attribute__((ext_vector_type(16))) __bf16 v16bf;
typedef __attribute__((ext_vector_type(8)))  __bf16 v8bf;
typedef __attribute__((ext_vector_type(4)))  __bf16 v4bf;
typedef __attribute__((ext_vector_type(8)))  float  v8f;

#define B_SZ   16384
#define N_IN   256
#define N_H    512

// LDS A-stripe layout: 16 rows, row stride K+8 bf16 (1040B for K=512) to
// de-conflict the 64-bank LDS on ds_load_b128 fragment reads.
#define LPAD   8
#define LK512  (N_H + LPAD)              // 520
#define LK256  (N_IN + LPAD)             // 264
#define SM_RD   0
#define SM_S0   (16 * LK256)             // 4224
#define SM_S1   (SM_S0 + 16 * LK512)     // 12544
#define SM_S2   (SM_S1 + 16 * LK512)     // 20864
#define SM_I0   (SM_S2 + 16 * LK512)     // 29184
#define SM_I1   (SM_I0 + 16 * LK512)     // 37504
#define SM_TOT  (SM_I1 + 16 * LK512)     // 45824 bf16 = 91648 B (~89.5 KB)

__device__ __forceinline__ float sigf(float x) {
  // sigmoid with HW transcendentals (v_exp_f32 + v_rcp_f32)
  return __builtin_amdgcn_rcpf(1.0f + __expf(-x));
}

// ---------------------------------------------------------------------------
struct CvtArgs {
  const float*    src[9];
  unsigned short* dst[9];
  int             n[9];
};

struct FusedArgs {
  const float* data;
  const float* s0; const float* s1; const float* s2;
  const float* i0; const float* i1;
  const float* tgt;
  const float* n_s0; const float* n_s1; const float* n_s2;
  const float* n_i0; const float* n_i1;
  const float* m1;  const float* m2;
  // W order: wpb0, wpi0, wpf0, wip0, wpb1, wpi1, wpf1, wip1, wpf2
  const unsigned short* W[9];
  float* out;
};

// ---------------- weight fp32 -> bf16 (tiny; lives in L2 afterwards) -------
__global__ void dn_cvt_weights(CvtArgs ca) {
  const int id = blockIdx.y;
  const int n  = ca.n[id];
  const float* __restrict__ src = ca.src[id];
  __bf16* __restrict__ dst = (__bf16*)ca.dst[id];
  for (int i = blockIdx.x * blockDim.x + threadIdx.x; i < n;
       i += gridDim.x * blockDim.x)
    dst[i] = (__bf16)src[i];
}

// ---------------- per-row mean of sigmoid(s1)/sigmoid(s2) ------------------
__global__ void dn_row_mean(const float* __restrict__ s1,
                            const float* __restrict__ s2,
                            float* __restrict__ m1, float* __restrict__ m2) {
  const float* src = blockIdx.y ? s2 : s1;
  float*       dst = blockIdx.y ? m2 : m1;
  const size_t base = (size_t)blockIdx.x * N_H;
  const int tid = threadIdx.x;                    // 256 threads, 8 wave32s
  float v = sigf(src[base + tid]) + sigf(src[base + tid + 256]);
#pragma unroll
  for (int off = 16; off > 0; off >>= 1) v += __shfl_down(v, off, 32);
  __shared__ float part[8];
  if ((tid & 31) == 0) part[tid >> 5] = v;
  __syncthreads();
  if (tid == 0) {
    float s = 0.f;
#pragma unroll
    for (int i = 0; i < 8; ++i) s += part[i];
    dst[blockIdx.x] = s * (1.0f / (float)N_H);
  }
}

// ---------------- cooperative sigmoid stage: 16 x K fp32 -> bf16 LDS -------
// Stripe is 16 consecutive full rows => one contiguous 16*K fp32 block.
template <int K>
__device__ __forceinline__ void stage_lds(const float* __restrict__ src,
                                          __bf16* dst, int rowbase, int tid) {
  constexpr int LK = K + LPAD;
  const float* g = src + (size_t)rowbase * K;
#pragma unroll
  for (int f = 0; f < 16 * K; f += 1024) {        // 256 thr x float4
    const int idx = f + tid * 4;
    const int row = idx / K;                      // K pow2 -> shifts
    const int col = idx % K;
    float4 a = *(const float4*)(g + idx);
    v4bf b;
    b[0] = (__bf16)sigf(a.x); b[1] = (__bf16)sigf(a.y);
    b[2] = (__bf16)sigf(a.z); b[3] = (__bf16)sigf(a.w);
    *(v4bf*)(dst + row * LK + col) = b;           // ds_store_b64
  }
}

// ---------------- WMMA GEMM chunk: 16 rows x 64 cols per wave --------------
// c[t][v] += sum_k ldsA[row,k] * W[col,k]   (A bf16 in LDS, W bf16 NxK in L2)
struct LaneInfo { int rowbase, colbase, half, l15; };

template <int K>
__device__ __forceinline__ void gemm_chunk(const __bf16* lA,
                                           const __bf16* __restrict__ Wb,
                                           const LaneInfo L, v8f c[4]) {
  constexpr int LK = K + LPAD;
  const v8f vzero = {};
#pragma unroll
  for (int t = 0; t < 4; ++t) c[t] = vzero;

  // ISA 16-bit A layout: lane = M&15; K = half*8+[0..7], then +16.
  const __bf16* arow = lA + L.l15 * LK + L.half * 8;
  const __bf16* wrow = Wb + (size_t)(L.colbase + L.l15) * K + L.half * 8;

  for (int k = 0; k < K; k += 32) {
    v8bf a0 = *(const v8bf*)(arow + k);           // ds_load_b128
    v8bf a1 = *(const v8bf*)(arow + k + 16);
    v16bf av;
#pragma unroll
    for (int e = 0; e < 8; ++e) { av[e] = a0[e]; av[e + 8] = a1[e]; }

#pragma unroll
    for (int t = 0; t < 4; ++t) {
      const __bf16* wp = wrow + (size_t)t * 16 * K + k;
      v8bf b0 = *(const v8bf*)(wp);               // global_load_b128 (L2)
      v8bf b1 = *(const v8bf*)(wp + 16);
      v16bf bv;
#pragma unroll
      for (int e = 0; e < 8; ++e) { bv[e] = b0[e]; bv[e + 8] = b1[e]; }
      c[t] = __builtin_amdgcn_wmma_f32_16x16x32_bf16(
          false, av, false, bv, (short)0, c[t], false, false);
    }
  }
}

// D layout: VGPR v -> row rowbase + half*8 + v, lane col = colbase + t*16 + l15
__device__ __forceinline__ size_t didx(const LaneInfo L, int t, int v) {
  return (size_t)(L.rowbase + L.half * 8 + v) * N_H + L.colbase + t * 16 + L.l15;
}

__device__ __forceinline__ void store_tiles(float* __restrict__ dst,
                                            const LaneInfo L, const v8f c[4]) {
#pragma unroll
  for (int t = 0; t < 4; ++t)
#pragma unroll
    for (int v = 0; v < 8; ++v) dst[didx(L, t, v)] = c[t][v];
}

// ---------------- one fused kernel: all 9 GEMMs + Euler epilogue -----------
__global__ __launch_bounds__(256) void dn_fused(FusedArgs fa) {
  __shared__ __align__(16) __bf16 smem[SM_TOT];

  const int tid = threadIdx.x;
  LaneInfo L;
  L.rowbase = blockIdx.x * 16;
  L.colbase = (tid >> 5) * 64;            // 8 waves cover N_H = 512
  L.half    = (tid >> 4) & 1;
  L.l15     = tid & 15;

  // ---- stage sigmoid(A) for all six A matrices once per block ----
  stage_lds<N_IN>(fa.data, smem + SM_RD, L.rowbase, tid);
  stage_lds<N_H>(fa.s0, smem + SM_S0, L.rowbase, tid);
  stage_lds<N_H>(fa.s1, smem + SM_S1, L.rowbase, tid);
  stage_lds<N_H>(fa.s2, smem + SM_S2, L.rowbase, tid);
  stage_lds<N_H>(fa.i0, smem + SM_I0, L.rowbase, tid);
  stage_lds<N_H>(fa.i1, smem + SM_I1, L.rowbase, tid);
  __syncthreads();

  constexpr float GLK = 0.1f, GB = 1.0f, GA = 0.8f, GD = 1.0f,
                  GSOM = 0.8f, NOISE = 0.1f, DT = 0.1f;
  const size_t S = (size_t)B_SZ * N_H;
  float* out = fa.out;

  v8f va[4], acc[4];

  // ================= layer 0 =================
  gemm_chunk<N_H>(smem + SM_S1, (const __bf16*)fa.W[0], L, va);   // va_td0
  store_tiles(out + 5 * S, L, va);
  gemm_chunk<N_H>(smem + SM_I0, (const __bf16*)fa.W[1], L, acc);  // va_c0
  store_tiles(out + 7 * S, L, acc);
#pragma unroll
  for (int t = 0; t < 4; ++t) va[t] += acc[t];                    // va0

  gemm_chunk<N_IN>(smem + SM_RD, (const __bf16*)fa.W[2], L, acc); // vb0
#pragma unroll
  for (int t = 0; t < 4; ++t)
#pragma unroll
    for (int v = 0; v < 8; ++v) {
      const size_t ix = didx(L, t, v);
      const float s = fa.s0[ix];
      out[ix] = s + DT * (-GLK * s + GB * (acc[t][v] - s) +
                          GA * (va[t][v] - s) + NOISE * fa.n_s0[ix]);
    }

  gemm_chunk<N_H>(smem + SM_S0, (const __bf16*)fa.W[3], L, acc);  // vi0
  {
    float mv[8];
#pragma unroll
    for (int v = 0; v < 8; ++v) mv[v] = fa.m1[L.rowbase + L.half * 8 + v];
#pragma unroll
    for (int t = 0; t < 4; ++t)
#pragma unroll
      for (int v = 0; v < 8; ++v) {
        const size_t ix = didx(L, t, v);
        const float iv = fa.i0[ix];
        out[3 * S + ix] = iv + DT * (-GLK * iv + GD * (acc[t][v] - iv) +
                                     GSOM * (mv[v] - iv) + NOISE * fa.n_i0[ix]);
      }
  }

  // ================= layer 1 =================
  gemm_chunk<N_H>(smem + SM_S2, (const __bf16*)fa.W[4], L, va);   // va_td1
  store_tiles(out + 6 * S, L, va);
  gemm_chunk<N_H>(smem + SM_I1, (const __bf16*)fa.W[5], L, acc);  // va_c1
  store_tiles(out + 8 * S, L, acc);
#pragma unroll
  for (int t = 0; t < 4; ++t) va[t] += acc[t];                    // va1

  gemm_chunk<N_H>(smem + SM_S0, (const __bf16*)fa.W[6], L, acc);  // vb1
#pragma unroll
  for (int t = 0; t < 4; ++t)
#pragma unroll
    for (int v = 0; v < 8; ++v) {
      const size_t ix = didx(L, t, v);
      const float s = fa.s1[ix];
      out[S + ix] = s + DT * (-GLK * s + GB * (acc[t][v] - s) +
                              GA * (va[t][v] - s) + NOISE * fa.n_s1[ix]);
    }

  gemm_chunk<N_H>(smem + SM_S1, (const __bf16*)fa.W[7], L, acc);  // vi1
  {
    float mv[8];
#pragma unroll
    for (int v = 0; v < 8; ++v) mv[v] = fa.m2[L.rowbase + L.half * 8 + v];
#pragma unroll
    for (int t = 0; t < 4; ++t)
#pragma unroll
      for (int v = 0; v < 8; ++v) {
        const size_t ix = didx(L, t, v);
        const float iv = fa.i1[ix];
        out[4 * S + ix] = iv + DT * (-GLK * iv + GD * (acc[t][v] - iv) +
                                     GSOM * (mv[v] - iv) + NOISE * fa.n_i1[ix]);
      }
  }

  // ================= output layer =================
  gemm_chunk<N_H>(smem + SM_S1, (const __bf16*)fa.W[8], L, acc);  // vb2
#pragma unroll
  for (int t = 0; t < 4; ++t)
#pragma unroll
    for (int v = 0; v < 8; ++v) {
      const size_t ix = didx(L, t, v);
      const float s = fa.s2[ix];
      out[2 * S + ix] = s + DT * (-GLK * s + GB * (acc[t][v] - s) +
                                  NOISE * fa.n_s2[ix] + GSOM * (fa.tgt[ix] - s));
    }
}

// ---------------------------------------------------------------------------
extern "C" void kernel_launch(void* const* d_in, const int* in_sizes, int n_in,
                              void* d_out, int out_size, void* d_ws, size_t ws_size,
                              hipStream_t stream) {
  (void)in_sizes; (void)n_in; (void)out_size; (void)ws_size;
  const float* in[21];
  for (int i = 0; i < 21; ++i) in[i] = (const float*)d_in[i];
  float* out = (float*)d_out;
  char*  ws  = (char*)d_ws;

  // fused W order: wpb0, wpi0, wpf0, wip0, wpb1, wpi1, wpf1, wip1, wpf2
  static const int wsrc[9] = {10, 14, 7, 12, 11, 15, 8, 13, 9};
  static const int kk[9]   = {N_H, N_H, N_IN, N_H, N_H, N_H, N_H, N_H, N_H};

  // workspace: bf16 weights (~4.4MB) + two per-row mean vectors
  size_t off = 0;
  unsigned short* wb[9];
  for (int m = 0; m < 9; ++m) {
    wb[m] = (unsigned short*)(ws + off);
    off += (size_t)N_H * kk[m] * sizeof(unsigned short);
  }
  float* m1 = (float*)(ws + off); off += (size_t)B_SZ * sizeof(float);
  float* m2 = (float*)(ws + off); off += (size_t)B_SZ * sizeof(float);

  CvtArgs ca;
  for (int m = 0; m < 9; ++m) {
    ca.src[m] = in[wsrc[m]]; ca.dst[m] = wb[m]; ca.n[m] = N_H * kk[m];
  }

  FusedArgs fa;
  fa.data = in[0];  fa.tgt = in[1];
  fa.s0 = in[2];  fa.s1 = in[3];  fa.s2 = in[4];
  fa.i0 = in[5];  fa.i1 = in[6];
  fa.n_s0 = in[16]; fa.n_s1 = in[17]; fa.n_s2 = in[18];
  fa.n_i0 = in[19]; fa.n_i1 = in[20];
  fa.m1 = m1; fa.m2 = m2;
  for (int m = 0; m < 9; ++m) fa.W[m] = wb[m];
  fa.out = out;

  dn_cvt_weights<<<dim3(1024, 9), 256, 0, stream>>>(ca);
  dn_row_mean<<<dim3(B_SZ, 2), 256, 0, stream>>>(in[3], in[4], m1, m2);
  dn_fused<<<dim3(B_SZ / 16), 256, 0, stream>>>(fa);
}
